// EmbeddingEMACE_36825049596263
// MI455X (gfx1250) — matrix operand order repaired and checked by reading, not compiled, not run on hardware
//
#include <hip/hip_runtime.h>
#include <hip/hip_bf16.h>
#include <math.h>

// ---------------------------------------------------------------------------
// MACE-like GNN forward for MI455X (gfx1250, wave32, WMMA).
// Edge MLP GEMMs fused + executed with v_wmma_f32_16x16x32_f16.
// ---------------------------------------------------------------------------

typedef __attribute__((ext_vector_type(16))) _Float16 v16h;
typedef __attribute__((ext_vector_type(8)))  float    v8f;

#define R_MAX 5.0f
#define AVG_NEIGH 16.0f
#define PI_F 3.14159265358979323846f

// Fast silu: x * rcp(1+exp(-x)) -- v_exp_f32 + v_add + v_rcp + v_mul,
// avoids the correctly-rounded v_div_scale/v_div_fmas expansion.
__device__ __forceinline__ float fast_rcp(float x) {
  return __builtin_amdgcn_rcpf(x);
}
__device__ __forceinline__ float silu_f(float x) {
  return x * fast_rcp(1.0f + __expf(-x));
}

// A-operand (16x32 f16, M=edge rows) per ISA 7.12.2: lanes 0-15 = M rows,
// VGPR v holds K = (v<4 ? 2v : 8+2v) + (lane>=16 ? 8 : 0), two halves per VGPR.
// buf is [16 rows][64 K] f16 (row stride 64); kc selects 32-wide K chunk.
__device__ __forceinline__ v16h load_A_tile(const _Float16* buf, int lane, int kc) {
  const int row  = lane & 15;
  const int half = lane >> 4;
  v16h a;
#pragma unroll
  for (int idx = 0; idx < 16; ++idx) {
    const int v  = idx >> 1;
    const int kl = ((v < 4) ? (2 * v) : (8 + 2 * v)) + (idx & 1) + half * 8;
    a[idx] = buf[row * 64 + kc * 32 + kl];
  }
  return a;
}

// B-operand (32x16 f16): lanes 0-15 hold N=lane with K=0..7,16..23,
// lanes 16-31 hold N=lane-16 with K=8..15,24..31. Weights staged [N][Kstride] f16.
__device__ __forceinline__ v16h load_B_tile(const _Float16* w, int lane, int nt,
                                            int kc, int kstride) {
  const int n     = nt * 16 + (lane & 15);
  const int khalf = lane >> 4;
  v16h b;
#pragma unroll
  for (int idx = 0; idx < 16; ++idx) {
    b[idx] = w[n * kstride + kc * 32 + khalf * 8 + idx + ((idx >= 8) ? 8 : 0)];
  }
  return b;
}

__device__ __forceinline__ v8f wmma_f16(v16h a, v16h b, v8f c) {
  return __builtin_amdgcn_wmma_f32_16x16x32_f16(false, a, false, b,
                                                (short)0, c, false, false);
}

// ---------------------------------------------------------------------------
// Utility kernels
// ---------------------------------------------------------------------------
__global__ void zero_f32_kernel(float* __restrict__ p, long long n) {
  long long i      = (long long)blockIdx.x * blockDim.x + threadIdx.x;
  long long stride = (long long)gridDim.x * blockDim.x;
  for (; i < n; i += stride) p[i] = 0.0f;
}

__global__ void copy_f32_kernel(const float* __restrict__ s, float* __restrict__ d, int n) {
  int i = blockIdx.x * blockDim.x + threadIdx.x;
  if (i < n) d[i] = s[i];
}

// ---------------------------------------------------------------------------
// Node init: h0 = node_attrs @ W_embed (one-hot => row select), e0 atomics.
// ---------------------------------------------------------------------------
__global__ void node_init_kernel(const float* __restrict__ node_attrs,
                                 const float* __restrict__ atomic_energies,
                                 const float* __restrict__ W_embed,
                                 const int* __restrict__ batch,
                                 float* __restrict__ h0,
                                 float* __restrict__ energies_acc,
                                 int* __restrict__ species, int N) {
  int idx = blockIdx.x * blockDim.x + threadIdx.x;
  if (idx >= N * 64) return;
  const int n = idx >> 6;
  const int c = idx & 63;
  // one-hot argmax
  int sp = 0;
  float best = -1e30f;
#pragma unroll
  for (int k = 0; k < 10; ++k) {
    float v = node_attrs[n * 10 + k];
    if (v > best) { best = v; sp = k; }
  }
  h0[idx] = W_embed[sp * 64 + c];
  if (c == 0) species[n] = sp;
  if (c < 3) atomicAdd(&energies_acc[batch[n] * 3 + c], atomic_energies[sp]);
}

// ---------------------------------------------------------------------------
// Edge geometry: spherical harmonics up to l=3 + Bessel radial basis.
// ---------------------------------------------------------------------------
__global__ void edge_geom_kernel(const float* __restrict__ pos,
                                 const float* __restrict__ shifts,
                                 const int* __restrict__ edge_index,
                                 float* __restrict__ Y, float* __restrict__ rb, int E) {
  int e = blockIdx.x * blockDim.x + threadIdx.x;
  if (e >= E) return;
  const int s = edge_index[e];
  const int r = edge_index[E + e];
  float vx = pos[r * 3 + 0] - pos[s * 3 + 0] + shifts[e * 3 + 0];
  float vy = pos[r * 3 + 1] - pos[s * 3 + 1] + shifts[e * 3 + 1];
  float vz = pos[r * 3 + 2] - pos[s * 3 + 2] + shifts[e * 3 + 2];
  float len = sqrtf(vx * vx + vy * vy + vz * vz + 1e-12f);
  float il = fast_rcp(len);
  float x = vx * il, y = vy * il, z = vz * il;

  const float s3 = 1.7320508075688772f;   // sqrt(3)
  const float s5 = 2.23606797749979f;     // sqrt(5)
  const float s15 = 3.872983346207417f;   // sqrt(15)
  const float c70 = 2.091650066335189f;   // sqrt(70)/4
  const float c105 = 10.246950765959598f; // sqrt(105)
  const float c42 = 1.6201851746019651f;  // sqrt(42)/4
  const float c7 = 1.3228756555322954f;   // sqrt(7)/2

  float* Ye = Y + (long long)e * 16;
  float xx = x * x, yy = y * y, zz = z * z;
  Ye[0]  = 1.0f;
  Ye[1]  = s3 * x;
  Ye[2]  = s3 * y;
  Ye[3]  = s3 * z;
  Ye[4]  = s15 * x * y;
  Ye[5]  = s15 * y * z;
  Ye[6]  = s5 * 0.5f * (3.0f * zz - 1.0f);
  Ye[7]  = s15 * x * z;
  Ye[8]  = s15 * 0.5f * (xx - yy);
  Ye[9]  = c70 * y * (3.0f * xx - yy);
  Ye[10] = c105 * x * y * z;
  Ye[11] = c42 * y * (5.0f * zz - 1.0f);
  Ye[12] = c7 * z * (5.0f * zz - 3.0f);
  Ye[13] = c42 * x * (5.0f * zz - 1.0f);
  Ye[14] = c105 * 0.5f * z * (xx - yy);
  Ye[15] = c70 * x * (xx - 3.0f * yy);

  float xr = len * (1.0f / R_MAX);
  float x5 = xr * xr * xr * xr * xr;
  float poly = 1.0f - 21.0f * x5 + 35.0f * x5 * xr - 15.0f * x5 * xr * xr;
  float cutoff = (xr < 1.0f) ? poly : 0.0f;
  float pref = sqrtf(2.0f / R_MAX) * il * cutoff;
  float* rbe = rb + (long long)e * 8;
#pragma unroll
  for (int k = 1; k <= 8; ++k) {
    rbe[k - 1] = pref * __sinf(PI_F * (float)k * xr);
  }
}

// ---------------------------------------------------------------------------
// Fused edge kernel: radial MLP (4 GEMMs via WMMA) + Y/h message + scatter.
// 128 threads = 4 waves; each wave owns a 16-edge tile with private LDS.
// ---------------------------------------------------------------------------
// LDS layout (dynamic):
//  w1 [64][32]  f16  4096 B   (Wr1 padded K 8->32)
//  w2 [64][64]  f16  8192 B
//  w3 [64][64]  f16  8192 B
//  w4 [256][64] f16 32768 B
//  per wave (x4): bufA[16][64]f16 2048, bufB[16][64]f16 2048,
//                 hsT[16][64]f32 4096, yT[16][16]f32 1024, snd/rcv 128
constexpr int SMEM_W_BYTES    = (64 * 32 + 64 * 64 + 64 * 64 + 256 * 64) * 2; // 53248
constexpr int SMEM_WAVE_BYTES = 16 * 64 * 2 * 2 + 16 * 64 * 4 + 16 * 16 * 4 + 16 * 4 * 2; // 9344
constexpr int SMEM_TOTAL      = SMEM_W_BYTES + 4 * SMEM_WAVE_BYTES; // 90624

__global__ __launch_bounds__(128)
void edge_layer_kernel(const float* __restrict__ h_cur,
                       const float* __restrict__ Y,
                       const float* __restrict__ rb,
                       const int* __restrict__ edge_index,
                       const float* __restrict__ Wr1, const float* __restrict__ Wr2,
                       const float* __restrict__ Wr3, const float* __restrict__ Wr4,
                       float* __restrict__ A_acc, int E) {
  extern __shared__ char smem[];
  _Float16* w1 = (_Float16*)smem;        // [64][32]
  _Float16* w2 = w1 + 64 * 32;           // [64][64]
  _Float16* w3 = w2 + 64 * 64;           // [64][64]
  _Float16* w4 = w3 + 64 * 64;           // [256][64]
  char* pw = (char*)(w4 + 256 * 64);

  const int wid  = threadIdx.x >> 5;
  const int lane = threadIdx.x & 31;
  char* my = pw + wid * SMEM_WAVE_BYTES;
  _Float16* bufA = (_Float16*)my;        // [16][64]
  _Float16* bufB = bufA + 16 * 64;       // [16][64]
  float* hsT = (float*)(bufB + 16 * 64); // [16][64]
  float* yT  = hsT + 16 * 64;            // [16][16]
  int* sndT  = (int*)(yT + 16 * 16);     // [16]
  int* rcvT  = sndT + 16;                // [16]

  // Stage weights into LDS in B-operand layout [N][K] f16 (cooperative).
  for (int i = threadIdx.x; i < 64 * 32; i += 128) {
    int n = i >> 5, k = i & 31;
    w1[i] = (_Float16)((k < 8) ? Wr1[k * 64 + n] : 0.0f);
  }
  for (int i = threadIdx.x; i < 64 * 64; i += 128) {
    int n = i >> 6, k = i & 63;
    w2[i] = (_Float16)Wr2[k * 64 + n];
    w3[i] = (_Float16)Wr3[k * 64 + n];
  }
  for (int i = threadIdx.x; i < 256 * 64; i += 128) {
    int n = i >> 6, k = i & 63;
    w4[i] = (_Float16)Wr4[k * 256 + n];
  }

  // Per-wave edge tile staging.
  const int et = (blockIdx.x * 4 + wid) * 16;
  if (lane < 16) {
    int e  = et + lane;
    int ec = (e < E) ? e : (E - 1);
    sndT[lane] = edge_index[ec];
    rcvT[lane] = edge_index[E + ec];
  }
  __syncthreads();

  for (int i = lane; i < 16 * 64; i += 32) {
    int row = i >> 6, c = i & 63;
    int e = et + row;
    int ec = (e < E) ? e : (E - 1);
    bufA[i] = (_Float16)((c < 8) ? rb[(long long)ec * 8 + c] : 0.0f);
    hsT[i]  = h_cur[(long long)sndT[row] * 64 + c];
  }
  for (int i = lane; i < 16 * 16; i += 32) {
    int row = i >> 4, lm = i & 15;
    int e = et + row;
    int ec = (e < E) ? e : (E - 1);
    yT[i] = Y[(long long)ec * 16 + lm];
  }
  __syncthreads();

  v8f acc[4];

  // --- GEMM1: (16x8 pad 32) @ (32x64), silu -> bufB ------------------------
#pragma unroll
  for (int nt = 0; nt < 4; ++nt) {
    v16h a = load_A_tile(bufA, lane, 0);
    v16h b = load_B_tile(w1, lane, nt, 0, 32);
    v8f c = {};
    acc[nt] = wmma_f16(a, b, c);
  }
#pragma unroll
  for (int nt = 0; nt < 4; ++nt) {
    const int col   = nt * 16 + (lane & 15);
    const int rbase = (lane >> 4) * 8;
#pragma unroll
    for (int j = 0; j < 8; ++j)
      bufB[(rbase + j) * 64 + col] = (_Float16)silu_f(acc[nt][j]);
  }
  __syncthreads();

  // --- GEMM2: (16x64) @ (64x64), silu -> bufA ------------------------------
#pragma unroll
  for (int nt = 0; nt < 4; ++nt) {
    v8f c = {};
#pragma unroll
    for (int kc = 0; kc < 2; ++kc) {
      v16h a = load_A_tile(bufB, lane, kc);
      v16h b = load_B_tile(w2, lane, nt, kc, 64);
      c = wmma_f16(a, b, c);
    }
    acc[nt] = c;
  }
  __syncthreads();
#pragma unroll
  for (int nt = 0; nt < 4; ++nt) {
    const int col   = nt * 16 + (lane & 15);
    const int rbase = (lane >> 4) * 8;
#pragma unroll
    for (int j = 0; j < 8; ++j)
      bufA[(rbase + j) * 64 + col] = (_Float16)silu_f(acc[nt][j]);
  }
  __syncthreads();

  // --- GEMM3: (16x64) @ (64x64), silu -> bufB ------------------------------
#pragma unroll
  for (int nt = 0; nt < 4; ++nt) {
    v8f c = {};
#pragma unroll
    for (int kc = 0; kc < 2; ++kc) {
      v16h a = load_A_tile(bufA, lane, kc);
      v16h b = load_B_tile(w3, lane, nt, kc, 64);
      c = wmma_f16(a, b, c);
    }
    acc[nt] = c;
  }
  __syncthreads();
#pragma unroll
  for (int nt = 0; nt < 4; ++nt) {
    const int col   = nt * 16 + (lane & 15);
    const int rbase = (lane >> 4) * 8;
#pragma unroll
    for (int j = 0; j < 8; ++j)
      bufB[(rbase + j) * 64 + col] = (_Float16)silu_f(acc[nt][j]);
  }
  __syncthreads();

  // --- GEMM4: (16x64) @ (64x256), fused message + scatter ------------------
  const float inv_avg = 1.0f / AVG_NEIGH;
  for (int nt = 0; nt < 16; ++nt) {
    v8f c = {};
#pragma unroll
    for (int kc = 0; kc < 2; ++kc) {
      v16h a = load_A_tile(bufB, lane, kc);
      v16h b = load_B_tile(w4, lane, nt, kc, 64);
      c = wmma_f16(a, b, c);
    }
    const int l   = nt >> 2;                 // which angular channel block
    const int col = (nt & 3) * 16 + (lane & 15);
    const int lm0 = l * l;
    const int lm1 = (l + 1) * (l + 1);
#pragma unroll
    for (int j = 0; j < 8; ++j) {
      const int m = (lane >> 4) * 8 + j;     // edge row within tile
      const int e = et + m;
      if (e < E) {
        const float rv   = c[j];
        const float base = hsT[m * 64 + col] * rv * inv_avg;
        const int   rcv  = rcvT[m];
        for (int lm = lm0; lm < lm1; ++lm) {
          atomicAdd(&A_acc[((long long)rcv * 16 + lm) * 64 + col],
                    base * yT[m * 16 + lm]);
        }
      }
    }
  }
}

// ---------------------------------------------------------------------------
// Node update: inv = A0 + sum A^2 ; h' = inv @ W_prod + h @ W_sc[sp] ; readout
// One 64-thread block per node.
// ---------------------------------------------------------------------------
__global__ __launch_bounds__(64)
void node_layer_kernel(const float* __restrict__ h_in,
                       const float* __restrict__ A_acc,
                       const float* __restrict__ W_sc,
                       const float* __restrict__ W_prod,
                       const float* __restrict__ W_read,
                       const int* __restrict__ species,
                       const int* __restrict__ batch,
                       float* __restrict__ h_out,
                       float* __restrict__ energies_acc, int N) {
  const int n = blockIdx.x;
  if (n >= N) return;
  const int c = threadIdx.x;
  __shared__ float invs[64];
  __shared__ float hsh[64];
  const float* An = A_acc + (long long)n * 16 * 64;
  float a0 = An[c];
  float ss = 0.0f;
#pragma unroll
  for (int lm = 0; lm < 16; ++lm) {
    float v = An[lm * 64 + c];
    ss += v * v;
  }
  invs[c] = a0 + ss;
  hsh[c]  = h_in[(long long)n * 64 + c];
  __syncthreads();

  const int sp = species[n];
  const float* Wsp = W_sc + (long long)sp * 64 * 64;
  float sc = 0.0f, hp = 0.0f;
#pragma unroll 8
  for (int k = 0; k < 64; ++k) {
    sc += hsh[k] * Wsp[k * 64 + c];
    hp += invs[k] * W_prod[k * 64 + c];
  }
  const float hn = hp + sc;
  h_out[(long long)n * 64 + c] = hn;

  const int g = batch[n];
#pragma unroll
  for (int j = 0; j < 3; ++j)
    atomicAdd(&energies_acc[g * 3 + j], hn * W_read[c * 3 + j]);
}

// ---------------------------------------------------------------------------
// Host-side launch orchestration
// ---------------------------------------------------------------------------
extern "C" void kernel_launch(void* const* d_in, const int* in_sizes, int n_in,
                              void* d_out, int out_size, void* d_ws, size_t ws_size,
                              hipStream_t stream) {
  const float* positions       = (const float*)d_in[0];
  const float* shifts          = (const float*)d_in[1];
  const float* node_attrs      = (const float*)d_in[2];
  const float* atomic_energies = (const float*)d_in[3];
  const float* W_embed         = (const float*)d_in[4];
  const float* Wr1             = (const float*)d_in[5];
  const float* Wr2             = (const float*)d_in[6];
  const float* Wr3             = (const float*)d_in[7];
  const float* Wr4             = (const float*)d_in[8];
  const float* W_sc            = (const float*)d_in[9];
  const float* W_prod          = (const float*)d_in[10];
  const float* W_read          = (const float*)d_in[11];
  const int*   edge_index      = (const int*)d_in[12];
  const int*   batch           = (const int*)d_in[13];

  const int N = in_sizes[0] / 3;   // positions (N,3)
  const int E = in_sizes[1] / 3;   // shifts (E,3)
  const int G = out_size / 3;      // energies (G,3)
  const int n_layers = in_sizes[5] / (8 * 64); // Wr1 (L,8,64)

  // Workspace layout (f32 elements)
  float* ws = (float*)d_ws;
  long long off = 0;
  float* h[3];
  h[0] = ws + off; off += (long long)N * 64;
  h[1] = ws + off; off += (long long)N * 64;
  h[2] = ws + off; off += (long long)N * 64;
  float* Yb   = ws + off; off += (long long)E * 16;
  float* rbb  = ws + off; off += (long long)E * 8;
  float* Aacc = ws + off; off += (long long)N * 16 * 64;
  float* eacc = ws + off; off += 64; // G*3 padded
  int*   spws = (int*)(ws + off);
  (void)ws_size; (void)n_in;

  // allow >64KB dynamic LDS for the fused edge kernel
  hipFuncSetAttribute((const void*)edge_layer_kernel,
                      hipFuncAttributeMaxDynamicSharedMemorySize, SMEM_TOTAL);

  // 1) zero energy accumulator
  zero_f32_kernel<<<1, 64, 0, stream>>>(eacc, (long long)G * 3);

  // 2) node embedding + e0
  {
    int total = N * 64;
    node_init_kernel<<<(total + 255) / 256, 256, 0, stream>>>(
        node_attrs, atomic_energies, W_embed, batch, h[0], eacc, spws, N);
  }

  // 3) edge geometry
  edge_geom_kernel<<<(E + 255) / 256, 256, 0, stream>>>(
      positions, shifts, edge_index, Yb, rbb, E);

  // 4) interaction layers
  for (int i = 0; i < n_layers; ++i) {
    zero_f32_kernel<<<2048, 256, 0, stream>>>(Aacc, (long long)N * 16 * 64);

    const int blocksE = (E + 63) / 64; // 4 waves x 16 edges per block
    edge_layer_kernel<<<blocksE, 128, SMEM_TOTAL, stream>>>(
        h[i], Yb, rbb, edge_index,
        Wr1 + (long long)i * 8 * 64,
        Wr2 + (long long)i * 64 * 64,
        Wr3 + (long long)i * 64 * 64,
        Wr4 + (long long)i * 64 * 256,
        Aacc, E);

    node_layer_kernel<<<N, 64, 0, stream>>>(
        h[i], Aacc,
        W_sc   + (long long)i * 10 * 64 * 64,
        W_prod + (long long)i * 64 * 64,
        W_read + (long long)i * 64 * 3,
        spws, batch, h[i + 1], eacc, N);
  }

  // 5) write out energies
  copy_f32_kernel<<<1, 64, 0, stream>>>(eacc, (float*)d_out, G * 3);
}
